// NerfRender_33371895890438
// MI455X (gfx1250) — matrix operand order repaired and checked by reading, not compiled, not run on hardware
//
#include <hip/hip_runtime.h>

// ---------------------------------------------------------------------------
// Fused NeRF renderer for gfx1250 (MI455X).
// 1 wave = 16 rays, loop over 64 samples. Five GEMMs per 16-point tile run on
// v_wmma_f32_16x16x32_f16 (48 WMMAs per tile). Weights live in LDS in
// B-fragment-friendly (N-major) f16 layout; activations staged via LDS.
// ---------------------------------------------------------------------------

typedef __attribute__((ext_vector_type(16))) _Float16 v16h;
typedef __attribute__((ext_vector_type(8)))  _Float16 v8h;
typedef __attribute__((ext_vector_type(8)))  float    v8f;

#define NB 16384      // rays
#define NS 64         // samples per ray
#define WPB 4         // waves per block
#define PIF 3.14159265358979323846f

// ---- dynamic LDS layout (bytes) -------------------------------------------
// weight tables (f16, N-major so B fragments are contiguous K runs)
#define OFF_W1T   0        // [64][64] : W1t[n][k]  = W1[k][n] (k<63 else 0)
#define OFF_W2T   8192     // [64][64] : W2t[n][k]  = W2[k][n]
#define OFF_W2R   16384    // [64][64] : (W2^T)t    = W2[n][k] (row major)
#define OFF_W1R   24576    // [64][64] : (W1^T)t    = W1[n][k] (n<63 else 0)
#define OFF_WC1T  32768    // [64][128]: Wc1t[n][k] = Wc1[k][n] (k<97 else 0)
#define OFF_B1    49152    // f32[64]
#define OFF_B2    49408    // f32[64]
#define OFF_BC1   49664    // f32[64]
#define OFF_WY    49920    // f32[64]
#define OFF_WC2   50176    // f32[64*3]
#define OFF_BC2   50944    // f32[3]
#define OFF_BY    50956    // f32[1]
#define WAVE_BASE 51200    // per-wave staging starts here
#define WAVE_BYTES 12288   // 8KB act staging + 4KB sorted-t
#define SMEM_TOTAL (WAVE_BASE + WPB * WAVE_BYTES)

// wave-local LDS ordering fence (DS ops are in-order within a wave)
#define WSYNC() asm volatile("s_wait_dscnt 0x0" ::: "memory")

#define WMMA(a, b, c) \
  __builtin_amdgcn_wmma_f32_16x16x32_f16(false, (a), false, (b), (short)0, (c), false, false)

// A fragment, 16x32 f16 tile from row-major LDS buffer (rowH halfs per row).
// Element e of lane L holds K = e + 8*((e>>3) + (L>>4)), M = L&15.
__device__ __forceinline__ v16h fragA(const _Float16* buf, int rowH, int kt, int lane) {
  const int m = lane & 15, g = lane >> 4;
  const _Float16* p = buf + m * rowH + kt * 32 + g * 8;
  v8h lo = *(const v8h*)p;
  v8h hi = *(const v8h*)(p + 16);
  return __builtin_shufflevector(lo, hi, 0,1,2,3,4,5,6,7,8,9,10,11,12,13,14,15);
}

// B fragment, 32x16 f16 tile from N-major weight table (rowH halfs per N row).
// Lane L holds column N = nt*16 + (L&15), K = e + 8*((e>>3) + (L>>4)) + kt*32.
__device__ __forceinline__ v16h fragB(const _Float16* w, int rowH, int nt, int kt, int lane) {
  const int n = lane & 15, g = lane >> 4;
  const _Float16* p = w + (nt * 16 + n) * rowH + kt * 32 + g * 8;
  v8h lo = *(const v8h*)p;
  v8h hi = *(const v8h*)(p + 16);
  return __builtin_shufflevector(lo, hi, 0,1,2,3,4,5,6,7,8,9,10,11,12,13,14,15);
}

__global__ void __launch_bounds__(32 * WPB)
nerf_render_kernel(const float* __restrict__ ray, const float* __restrict__ tin,
                   const float* __restrict__ W1,  const float* __restrict__ b1,
                   const float* __restrict__ W2,  const float* __restrict__ b2,
                   const float* __restrict__ Wy,  const float* __restrict__ by,
                   const float* __restrict__ Wc1, const float* __restrict__ bc1,
                   const float* __restrict__ Wc2, const float* __restrict__ bc2,
                   float* __restrict__ out) {
  extern __shared__ char smem[];
  _Float16* W1t  = (_Float16*)(smem + OFF_W1T);
  _Float16* W2t  = (_Float16*)(smem + OFF_W2T);
  _Float16* W2r  = (_Float16*)(smem + OFF_W2R);
  _Float16* W1r  = (_Float16*)(smem + OFF_W1R);
  _Float16* Wc1t = (_Float16*)(smem + OFF_WC1T);
  float* b1L  = (float*)(smem + OFF_B1);
  float* b2L  = (float*)(smem + OFF_B2);
  float* bc1L = (float*)(smem + OFF_BC1);
  float* WyL  = (float*)(smem + OFF_WY);
  float* Wc2L = (float*)(smem + OFF_WC2);
  float* bc2L = (float*)(smem + OFF_BC2);
  float* byL  = (float*)(smem + OFF_BY);

  const int tid = threadIdx.x;
  // -------- stage weights into LDS (f16, B-fragment layouts) --------
  for (int i = tid; i < 64 * 64; i += 32 * WPB) {
    const int k = i >> 6, n = i & 63;
    W1t[n * 64 + k] = (_Float16)((k < 63) ? W1[k * 64 + n] : 0.f);
    W2t[n * 64 + k] = (_Float16)W2[k * 64 + n];
    W2r[n * 64 + k] = (_Float16)W2[n * 64 + k];
    W1r[n * 64 + k] = (_Float16)((n < 63) ? W1[n * 64 + k] : 0.f);
  }
  for (int i = tid; i < 64 * 128; i += 32 * WPB) {
    const int n = i >> 7, k = i & 127;
    Wc1t[n * 128 + k] = (_Float16)((k < 97) ? Wc1[k * 64 + n] : 0.f);
  }
  if (tid < 64) { b1L[tid] = b1[tid]; b2L[tid] = b2[tid]; bc1L[tid] = bc1[tid]; WyL[tid] = Wy[tid]; }
  for (int i = tid; i < 192; i += 32 * WPB) Wc2L[i] = Wc2[i];
  if (tid == 0) { bc2L[0] = bc2[0]; bc2L[1] = bc2[1]; bc2L[2] = bc2[2]; byL[0] = by[0]; }
  __syncthreads();

  const int lane = tid & 31, wave = tid >> 5;
  char* wb = smem + WAVE_BASE + wave * WAVE_BYTES;
  _Float16* bufA = (_Float16*)wb;             // 16x64 f16 (pe, dz2)
  _Float16* bufB = (_Float16*)(wb + 2048);    // 16x64 f16 (a1, dz1)
  _Float16* bufC = (_Float16*)(wb + 4096);    // 16x128 f16 (cin, h in cols 6..69)
  float*    dpeB = (float*)wb;                // 16x64 f32 (reuses bufA+bufB)
  _Float16* hcB  = (_Float16*)wb;             // 16x64 f16 (color hidden, reuses bufA)
  float*    tS   = (float*)(wb + 8192);       // 16x64 f32 sorted t

  const int rayBase = (blockIdx.x * WPB + wave) * 16;

  // -------- load + sort t per ray (one lane per ray, LDS insertion sort) ----
  for (int i = lane; i < 16 * 64; i += 32)
    tS[i] = tin[(rayBase + (i >> 6)) * 64 + (i & 63)];
  WSYNC();
  float o0 = 0, o1 = 0, o2 = 0, d0 = 0, d1 = 0, d2 = 0;
  if (lane < 16) {
    const float* rp = ray + (rayBase + lane) * 6;
    o0 = rp[0]; o1 = rp[1]; o2 = rp[2]; d0 = rp[3]; d1 = rp[4]; d2 = rp[5];
    float* tr = tS + lane * 64;
    for (int i = 1; i < 64; ++i) {
      float key = tr[i]; int j = i - 1;
      while (j >= 0 && tr[j] > key) { tr[j + 1] = tr[j]; --j; }
      tr[j + 1] = key;
    }
  }
  WSYNC();

  const float FR[10] = {PIF, 2*PIF, 4*PIF, 8*PIF, 16*PIF, 32*PIF, 64*PIF, 128*PIF, 256*PIF, 512*PIF};
  float T = 1.f, accR = 0.f, accG = 0.f, accBl = 0.f, accD = 0.f;

  for (int s = 0; s < 64; ++s) {
    WSYNC();  // previous iteration's hc reads precede this iteration's pe writes
    float tv = 0.f, x0 = 0.f, x1 = 0.f, x2 = 0.f;
    // -------- posenc(x) into bufA --------
    if (lane < 16) {
      tv = tS[lane * 64 + s];
      x0 = o0 + tv * d0; x1 = o1 + tv * d1; x2 = o2 + tv * d2;
      _Float16* pr = bufA + lane * 64;
      pr[0] = (_Float16)x0; pr[1] = (_Float16)x1; pr[2] = (_Float16)x2; pr[63] = (_Float16)0.f;
      const float xs[3] = {x0, x1, x2};
#pragma unroll
      for (int j = 0; j < 3; ++j)
#pragma unroll
        for (int l = 0; l < 10; ++l) {
          const float a = xs[j] * FR[l];
          pr[3  + j * 10 + l] = (_Float16)__sinf(a);
          pr[33 + j * 10 + l] = (_Float16)__cosf(a);
        }
    }
    WSYNC();

    const int col = lane & 15, mb = (lane >> 4) * 8;
    v8f acc1[4];  // a1 (post-ReLU) kept in registers for the backward mask

    // -------- L1: a1 = relu(pe @ W1 + b1) -> bufB --------
    {
      v16h a0 = fragA(bufA, 64, 0, lane), a1v = fragA(bufA, 64, 1, lane);
#pragma unroll
      for (int nt = 0; nt < 4; ++nt) {
        v8f c = {};
        c = WMMA(a0,  fragB(W1t, 64, nt, 0, lane), c);
        c = WMMA(a1v, fragB(W1t, 64, nt, 1, lane), c);
        const int n0 = nt * 16 + col; const float bb = b1L[n0];
#pragma unroll
        for (int v = 0; v < 8; ++v) {
          const float z = fmaxf(c[v] + bb, 0.f);
          acc1[nt][v] = z;
          bufB[(mb + v) * 64 + n0] = (_Float16)z;
        }
      }
    }
    WSYNC();

    // -------- L2: h = relu(a1 @ W2 + b2) -> bufC[:,6:70]; dz2 = (h>0)*Wy -> bufA
    {
      v16h a0 = fragA(bufB, 64, 0, lane), a1v = fragA(bufB, 64, 1, lane);
#pragma unroll
      for (int nt = 0; nt < 4; ++nt) {
        v8f c = {};
        c = WMMA(a0,  fragB(W2t, 64, nt, 0, lane), c);
        c = WMMA(a1v, fragB(W2t, 64, nt, 1, lane), c);
        const int n0 = nt * 16 + col; const float bb = b2L[n0], wy = WyL[n0];
#pragma unroll
        for (int v = 0; v < 8; ++v) {
          const float h = fmaxf(c[v] + bb, 0.f);
          bufC[(mb + v) * 128 + 6 + n0] = (_Float16)h;
          bufA[(mb + v) * 64 + n0] = (_Float16)((h > 0.f) ? wy : 0.f);
        }
      }
    }
    WSYNC();

    // -------- B1: dz1 = relu'(z1) * (dz2 @ W2^T) -> bufB --------
    {
      v16h a0 = fragA(bufA, 64, 0, lane), a1v = fragA(bufA, 64, 1, lane);
#pragma unroll
      for (int nt = 0; nt < 4; ++nt) {
        v8f c = {};
        c = WMMA(a0,  fragB(W2r, 64, nt, 0, lane), c);
        c = WMMA(a1v, fragB(W2r, 64, nt, 1, lane), c);
        const int n0 = nt * 16 + col;
#pragma unroll
        for (int v = 0; v < 8; ++v)
          bufB[(mb + v) * 64 + n0] = (_Float16)((acc1[nt][v] > 0.f) ? c[v] : 0.f);
      }
    }
    WSYNC();

    // -------- B2: dpe = dz1 @ W1^T -> dpeB (f32, reuses bufA+bufB space) ----
    {
      v16h a0 = fragA(bufB, 64, 0, lane), a1v = fragA(bufB, 64, 1, lane);
#pragma unroll
      for (int nt = 0; nt < 4; ++nt) {
        v8f c = {};
        c = WMMA(a0,  fragB(W1r, 64, nt, 0, lane), c);
        c = WMMA(a1v, fragB(W1r, 64, nt, 1, lane), c);
        const int n0 = nt * 16 + col;
#pragma unroll
        for (int v = 0; v < 8; ++v) dpeB[(mb + v) * 64 + n0] = c[v];
      }
    }
    WSYNC();

    // -------- per-point: normal, sigma, assemble cin extras --------
    float sigma = 0.f;
    if (lane < 16) {
      const float* dp = dpeB + lane * 64;
      const float xs[3] = {x0, x1, x2};
      float g[3];
#pragma unroll
      for (int j = 0; j < 3; ++j) {
        float gj = dp[j];
#pragma unroll
        for (int l = 0; l < 10; ++l) {
          const float f = FR[l], a = xs[j] * f;
          gj += f * (__cosf(a) * dp[3 + j * 10 + l] - __sinf(a) * dp[33 + j * 10 + l]);
        }
        g[j] = gj;
      }
      const float inv = 1.f / (sqrtf(g[0]*g[0] + g[1]*g[1] + g[2]*g[2]) + 1e-6f);
      _Float16* hr = bufC + lane * 128;
      float y = byL[0];
#pragma unroll
      for (int k = 0; k < 64; ++k) y += (float)hr[6 + k] * WyL[k];
      sigma = 1.f / (1.f + __expf(10.f * y));           // sigmoid(-10*y)
      hr[0] = (_Float16)x0; hr[1] = (_Float16)x1; hr[2] = (_Float16)x2;
      hr[3] = (_Float16)(g[0]*inv); hr[4] = (_Float16)(g[1]*inv); hr[5] = (_Float16)(g[2]*inv);
      const float ds[3] = {d0, d1, d2};
#pragma unroll
      for (int j = 0; j < 3; ++j) {
        hr[70 + j] = (_Float16)ds[j];
#pragma unroll
        for (int l = 0; l < 4; ++l) {
          const float a = ds[j] * FR[l];
          hr[73 + j * 4 + l] = (_Float16)__sinf(a);
          hr[85 + j * 4 + l] = (_Float16)__cosf(a);
        }
      }
#pragma unroll
      for (int k = 97; k < 128; ++k) hr[k] = (_Float16)0.f;
    }
    WSYNC();

    // -------- C1: hc = relu(cin @ Wc1 + bc1) -> hcB --------
    {
      v16h a0 = fragA(bufC, 128, 0, lane), a1v = fragA(bufC, 128, 1, lane);
      v16h a2 = fragA(bufC, 128, 2, lane), a3v = fragA(bufC, 128, 3, lane);
#pragma unroll
      for (int nt = 0; nt < 4; ++nt) {
        v8f c = {};
        c = WMMA(a0,  fragB(Wc1t, 128, nt, 0, lane), c);
        c = WMMA(a1v, fragB(Wc1t, 128, nt, 1, lane), c);
        c = WMMA(a2,  fragB(Wc1t, 128, nt, 2, lane), c);
        c = WMMA(a3v, fragB(Wc1t, 128, nt, 3, lane), c);
        const int n0 = nt * 16 + col; const float bb = bc1L[n0];
#pragma unroll
        for (int v = 0; v < 8; ++v)
          hcB[(mb + v) * 64 + n0] = (_Float16)fmaxf(c[v] + bb, 0.f);
      }
    }
    WSYNC();

    // -------- color head + compositing (per ray lane) --------
    if (lane < 16) {
      const _Float16* hc = hcB + lane * 64;
      float c0 = bc2L[0], c1 = bc2L[1], c2 = bc2L[2];
#pragma unroll
      for (int k = 0; k < 64; ++k) {
        const float hv = (float)hc[k];
        c0 += hv * Wc2L[k * 3 + 0];
        c1 += hv * Wc2L[k * 3 + 1];
        c2 += hv * Wc2L[k * 3 + 2];
      }
      c0 = 1.f / (1.f + __expf(-c0));
      c1 = 1.f / (1.f + __expf(-c1));
      c2 = 1.f / (1.f + __expf(-c2));
      const float wi = sigma * T;
      accR += wi * c0; accG += wi * c1; accBl += wi * c2; accD += wi * tv;
      out[NB * 4 + (rayBase + lane) * 64 + s] = wi;   // wi block after rgb+depth
      T *= (1.f - sigma);
    }
  }

  if (lane < 16) {
    const int r = rayBase + lane;
    out[r * 3 + 0] = accR;
    out[r * 3 + 1] = accG;
    out[r * 3 + 2] = accBl;
    out[NB * 3 + r] = accD;
  }
}

extern "C" void kernel_launch(void* const* d_in, const int* in_sizes, int n_in,
                              void* d_out, int out_size, void* d_ws, size_t ws_size,
                              hipStream_t stream) {
  (void)in_sizes; (void)n_in; (void)d_ws; (void)ws_size; (void)out_size;
  const float* ray = (const float*)d_in[0];
  const float* t   = (const float*)d_in[1];
  const float* W1  = (const float*)d_in[2];
  const float* b1  = (const float*)d_in[3];
  const float* W2  = (const float*)d_in[4];
  const float* b2  = (const float*)d_in[5];
  const float* Wy  = (const float*)d_in[6];
  const float* by  = (const float*)d_in[7];
  const float* Wc1 = (const float*)d_in[8];
  const float* bc1 = (const float*)d_in[9];
  const float* Wc2 = (const float*)d_in[10];
  const float* bc2 = (const float*)d_in[11];
  float* out = (float*)d_out;

  const int blocks = NB / (16 * WPB);  // 256 blocks x 128 threads (4 waves)
  nerf_render_kernel<<<blocks, 32 * WPB, SMEM_TOTAL, stream>>>(
      ray, t, W1, b1, W2, b2, Wy, by, Wc1, bc1, Wc2, bc2, out);
}
